// LSTMDetector_41059887350258
// MI455X (gfx1250) — compile-verified, matrix operand order checked by reading, not run on hardware
//
#include <hip/hip_runtime.h>
#include <hip/hip_fp16.h>

// LSTMDetector for MI455X (gfx1250): persistent 16-workgroup kernel.
// WGs 0..7  : layer-0, hidden slice [32w, 32w+32)  (weights Whh0 slice in LDS)
// WGs 8..15 : layer-1, hidden slice [32(w-8), ...)  (Wih1 + Whh1 slices in LDS)
// Pipelined: at iteration k, layer-0 computes step t=k, layer-1 computes t=k-1.
// h1/h2 exchanged via ping-pong fp16 buffers in d_ws; per-step L2 atomic barrier.
// GEMMs use v_wmma_f32_16x16x32_f16; c-state lives in VGPRs for all 2048 steps.

#define HID   256
#define BB    128
#define TT    2048
#define NWG   16
#define NL0   8
#define SP    264   // padded LDS row stride (f16 elems) for h / weight tiles
#define SPF   132   // padded LDS row stride (f32 elems) for gate pre-acts
#define PADV  (-100.0f)

typedef __attribute__((ext_vector_type(16))) _Float16 v16h;
typedef __attribute__((ext_vector_type(8)))  _Float16 v8h;
typedef __attribute__((ext_vector_type(8)))  float    v8f;

union AFrag { v16h v; v8h h[2]; };

__device__ __forceinline__ float sigm(float x)      { return 1.0f / (1.0f + __expf(-x)); }
__device__ __forceinline__ float tanh_fast(float x) { return 2.0f / (1.0f + __expf(-2.0f*x)) - 1.0f; }

// ws layout: [0]=cnt u32, [4]=gen u32, pad to 256B,
//            h1 ping-pong 2*BB*HID f16, h2 ping-pong 2*BB*HID f16.

__global__ void lstm_init(float* __restrict__ out, const float* __restrict__ b_fc,
                          int outN, unsigned* __restrict__ hdr,
                          _Float16* __restrict__ hbufs, int hHalves) {
    int i = blockIdx.x * blockDim.x + threadIdx.x;
    int stride = gridDim.x * blockDim.x;
    if (i < 2) hdr[i] = 0u;
    for (int j = i; j < outN; j += stride) out[j] = b_fc[j & 1];
    for (int j = i; j < hHalves; j += stride) hbufs[j] = (_Float16)0.0f;
}

__global__ __launch_bounds__(256)
void lstm_persistent(const float* __restrict__ y,
                     const float* __restrict__ W_ih0, const float* __restrict__ W_hh0,
                     const float* __restrict__ b_ih0, const float* __restrict__ b_hh0,
                     const float* __restrict__ W_ih1, const float* __restrict__ W_hh1,
                     const float* __restrict__ b_ih1, const float* __restrict__ b_hh1,
                     const float* __restrict__ W_fc,  const float* __restrict__ b_fc,
                     float* __restrict__ out, char* __restrict__ ws) {
    extern __shared__ char smem[];
    _Float16* WBa = (_Float16*)smem;            // 128 x SP  (L0: Whh0 slice, L1: Wih1 slice)
    _Float16* WBb = WBa + 128 * SP;             // 128 x SP  (L1: Whh1 slice)
    _Float16* HA1 = WBb + 128 * SP;             // 128 x SP  h1 stage; aliased by PRE
    _Float16* HA2 = HA1 + 128 * SP;             // 128 x SP  h2 stage (L1 only)
    float*    PRE  = (float*)HA1;               // 128 x SPF f32 gate pre-activations
    float*    ywin = (float*)(HA2 + 128 * SP);  // 4 x BB rolling input window

    const int wg    = blockIdx.x;
    const bool isL0 = (wg < NL0);
    const int slice = isL0 ? wg : (wg - NL0);   // hidden units [32*slice, 32*slice+32)
    const int tid   = threadIdx.x;
    const int lane  = tid & 31;
    const int wave  = tid >> 5;

    unsigned* cnt = (unsigned*)ws;
    unsigned* gen = cnt + 1;
    _Float16* h1buf = (_Float16*)(ws + 256);
    _Float16* h2buf = h1buf + 2 * BB * HID;

    // ---- Prologue: convert this WG's weight slice fp32 -> fp16 into LDS ----
    {
        int r    = tid >> 1;                 // local gate row 0..127
        int k0   = (tid & 1) * 128;
        int gate = r >> 5;
        int R    = gate * HID + slice * 32 + (r & 31);   // global gate row
        const float* s0 = (isL0 ? W_hh0 : W_ih1) + (size_t)R * HID + k0;
        for (int k = 0; k < 128; ++k) WBa[r * SP + k0 + k] = (_Float16)s0[k];
        if (!isL0) {
            const float* s1 = W_hh1 + (size_t)R * HID + k0;
            for (int k = 0; k < 128; ++k) WBb[r * SP + k0 + k] = (_Float16)s1[k];
        }
    }
    // ---- Per-thread update-phase constants (thread tid -> unit uu, batch group) ----
    const int uu   = tid & 31;     // hidden-unit offset within slice
    const int bgrp = tid >> 5;     // 8 groups of 16 batch rows
    float biasr[4], wihreg[4][4];
    float wfc0 = 0.0f, wfc1 = 0.0f;
    {
        int ug = slice * 32 + uu;
        for (int g4 = 0; g4 < 4; ++g4) {
            int R = g4 * HID + ug;
            biasr[g4] = isL0 ? (b_ih0[R] + b_hh0[R]) : (b_ih1[R] + b_hh1[R]);
            if (isL0) for (int j = 0; j < 4; ++j) wihreg[g4][j] = W_ih0[R * 4 + j];
            else for (int j = 0; j < 4; ++j) wihreg[g4][j] = 0.0f;
        }
        if (!isL0) { wfc0 = W_fc[ug]; wfc1 = W_fc[HID + ug]; }
    }
    if (isL0 && tid < BB) {        // padding columns for t = -3..-1
        ywin[1 * BB + tid] = PADV; ywin[2 * BB + tid] = PADV; ywin[3 * BB + tid] = PADV;
    }
    float cst[16];
    #pragma unroll
    for (int i = 0; i < 16; ++i) cst[i] = 0.0f;

    // GEMM tile ownership: wave -> 2 M-tiles x 4 N-tiles
    const int mt0    = (wave >> 1) * 2;
    const int ntbase = (wave & 1) * 4;
    const int l15    = lane & 15;
    const int khalf  = (lane >> 4) * 8;

    for (int k = 0; k <= TT; ++k) {
        const bool act = isL0 ? (k < TT) : (k >= 1);
        const int  t   = isL0 ? k : (k - 1);

        // ---- Stage h (and input column) into LDS ----
        if (act) {
            int b = tid >> 1, k0 = (tid & 1) * 128;
            if (isL0) {
                const _Float16* s = h1buf + ((k + 1) & 1) * (BB * HID) + b * HID + k0;
                _Float16* d = HA1 + b * SP + k0;
                #pragma unroll
                for (int i = 0; i < 128; i += 8) *(v8h*)(d + i) = *(const v8h*)(s + i);
                if (tid < BB) ywin[(k & 3) * BB + tid] = y[tid * TT + k];
            } else {
                const _Float16* s1 = h1buf + ((k - 1) & 1) * (BB * HID) + b * HID + k0;
                const _Float16* s2 = h2buf + (k & 1) * (BB * HID) + b * HID + k0;
                _Float16* d1 = HA1 + b * SP + k0;
                _Float16* d2 = HA2 + b * SP + k0;
                #pragma unroll
                for (int i = 0; i < 128; i += 8) {
                    *(v8h*)(d1 + i) = *(const v8h*)(s1 + i);
                    *(v8h*)(d2 + i) = *(const v8h*)(s2 + i);
                }
            }
        }
        __syncthreads();

        // ---- WMMA GEMM: G = h_prev x W^T (L1 adds second source) ----
        v8f acc[2][4];
        if (act) {
            #pragma unroll
            for (int mi = 0; mi < 2; ++mi)
                #pragma unroll
                for (int ni = 0; ni < 4; ++ni)
                    acc[mi][ni] = (v8f){0,0,0,0,0,0,0,0};
            for (int kt = 0; kt < 8; ++kt) {
                const int kb = kt * 32 + khalf;
                AFrag a0, a1, bf;
                a0.h[0] = *(const v8h*)&HA1[(mt0 * 16 + l15) * SP + kb];
                a0.h[1] = *(const v8h*)&HA1[(mt0 * 16 + l15) * SP + kb + 16];
                a1.h[0] = *(const v8h*)&HA1[((mt0 + 1) * 16 + l15) * SP + kb];
                a1.h[1] = *(const v8h*)&HA1[((mt0 + 1) * 16 + l15) * SP + kb + 16];
                #pragma unroll
                for (int ni = 0; ni < 4; ++ni) {
                    const int col = (ntbase + ni) * 16 + l15;
                    bf.h[0] = *(const v8h*)&WBa[col * SP + kb];
                    bf.h[1] = *(const v8h*)&WBa[col * SP + kb + 16];
                    acc[0][ni] = __builtin_amdgcn_wmma_f32_16x16x32_f16(
                        false, a0.v, false, bf.v, (short)0, acc[0][ni], false, false);
                    acc[1][ni] = __builtin_amdgcn_wmma_f32_16x16x32_f16(
                        false, a1.v, false, bf.v, (short)0, acc[1][ni], false, false);
                }
                if (!isL0) {
                    a0.h[0] = *(const v8h*)&HA2[(mt0 * 16 + l15) * SP + kb];
                    a0.h[1] = *(const v8h*)&HA2[(mt0 * 16 + l15) * SP + kb + 16];
                    a1.h[0] = *(const v8h*)&HA2[((mt0 + 1) * 16 + l15) * SP + kb];
                    a1.h[1] = *(const v8h*)&HA2[((mt0 + 1) * 16 + l15) * SP + kb + 16];
                    #pragma unroll
                    for (int ni = 0; ni < 4; ++ni) {
                        const int col = (ntbase + ni) * 16 + l15;
                        bf.h[0] = *(const v8h*)&WBb[col * SP + kb];
                        bf.h[1] = *(const v8h*)&WBb[col * SP + kb + 16];
                        acc[0][ni] = __builtin_amdgcn_wmma_f32_16x16x32_f16(
                            false, a0.v, false, bf.v, (short)0, acc[0][ni], false, false);
                        acc[1][ni] = __builtin_amdgcn_wmma_f32_16x16x32_f16(
                            false, a1.v, false, bf.v, (short)0, acc[1][ni], false, false);
                    }
                }
            }
        }
        __syncthreads();   // GEMM reads of HA1 done before PRE (alias) is written

        // ---- Scatter pre-activations to LDS ----
        if (act) {
            #pragma unroll
            for (int mi = 0; mi < 2; ++mi)
                #pragma unroll
                for (int ni = 0; ni < 4; ++ni) {
                    const int brow = (mt0 + mi) * 16 + (lane >> 4) * 8;
                    const int col  = (ntbase + ni) * 16 + l15;
                    #pragma unroll
                    for (int r = 0; r < 8; ++r)
                        PRE[(brow + r) * SPF + col] = acc[mi][ni][r];
                }
        }
        __syncthreads();

        // ---- Gate nonlinearity + state update (c in VGPRs) ----
        if (act) {
            const int ug = slice * 32 + uu;
            for (int bi = 0; bi < 16; ++bi) {
                const int b = bgrp * 16 + bi;
                float gi = PRE[b * SPF + uu]        + biasr[0];
                float gf = PRE[b * SPF + 32 + uu]   + biasr[1];
                float gg = PRE[b * SPF + 64 + uu]   + biasr[2];
                float go = PRE[b * SPF + 96 + uu]   + biasr[3];
                if (isL0) {
                    #pragma unroll
                    for (int j = 0; j < 4; ++j) {
                        const float xv = ywin[((t + j - 3) & 3) * BB + b];
                        gi += xv * wihreg[0][j]; gf += xv * wihreg[1][j];
                        gg += xv * wihreg[2][j]; go += xv * wihreg[3][j];
                    }
                }
                const float c = sigm(gf) * cst[bi] + sigm(gi) * tanh_fast(gg);
                cst[bi] = c;
                const float h = sigm(go) * tanh_fast(c);
                if (isL0) {
                    h1buf[(k & 1) * (BB * HID) + b * HID + ug] = (_Float16)h;
                } else {
                    h2buf[((k - 1) & 1) * (BB * HID) + b * HID + ug] = (_Float16)h;
                    float v0 = h * wfc0, v1 = h * wfc1;
                    #pragma unroll
                    for (int m = 16; m >= 1; m >>= 1) {
                        v0 += __shfl_xor(v0, m, 32);
                        v1 += __shfl_xor(v1, m, 32);
                    }
                    if (lane == 0) {
                        atomicAdd(out + ((size_t)b * TT + t) * 2 + 0, v0);
                        atomicAdd(out + ((size_t)b * TT + t) * 2 + 1, v1);
                    }
                }
            }
        }

        // ---- Device-scope generation barrier across the 16 persistent WGs ----
        __syncthreads();
        if (tid == 0) {
            __threadfence();
            unsigned g = __hip_atomic_load(gen, __ATOMIC_RELAXED, __HIP_MEMORY_SCOPE_AGENT);
            unsigned a = atomicAdd(cnt, 1u);
            if (a == NWG - 1) {
                __hip_atomic_store(cnt, 0u, __ATOMIC_RELAXED, __HIP_MEMORY_SCOPE_AGENT);
                __hip_atomic_store(gen, g + 1u, __ATOMIC_RELEASE, __HIP_MEMORY_SCOPE_AGENT);
            } else {
                while (__hip_atomic_load(gen, __ATOMIC_ACQUIRE, __HIP_MEMORY_SCOPE_AGENT) == g)
                    __builtin_amdgcn_s_sleep(1);
            }
            __threadfence();
        }
        __syncthreads();
    }
}

extern "C" void kernel_launch(void* const* d_in, const int* in_sizes, int n_in,
                              void* d_out, int out_size, void* d_ws, size_t ws_size,
                              hipStream_t stream) {
    const float* y     = (const float*)d_in[0];
    const float* W_ih0 = (const float*)d_in[1];
    const float* W_hh0 = (const float*)d_in[2];
    const float* b_ih0 = (const float*)d_in[3];
    const float* b_hh0 = (const float*)d_in[4];
    const float* W_ih1 = (const float*)d_in[5];
    const float* W_hh1 = (const float*)d_in[6];
    const float* b_ih1 = (const float*)d_in[7];
    const float* b_hh1 = (const float*)d_in[8];
    const float* W_fc  = (const float*)d_in[9];
    const float* b_fc  = (const float*)d_in[10];
    float* out = (float*)d_out;
    char*  ws  = (char*)d_ws;

    // Init: barrier header = 0, h ping-pong buffers = 0, out = broadcast fc bias.
    lstm_init<<<1024, 256, 0, stream>>>(out, b_fc, out_size,
                                        (unsigned*)ws,
                                        (_Float16*)(ws + 256), 4 * BB * HID);

    const size_t shmem = (size_t)(4 * 128 * SP) * sizeof(_Float16)   // WBa, WBb, HA1, HA2
                       + (size_t)(4 * BB) * sizeof(float);           // ywin
    lstm_persistent<<<NWG, 256, shmem, stream>>>(y, W_ih0, W_hh0, b_ih0, b_hh0,
                                                 W_ih1, W_hh1, b_ih1, b_hh1,
                                                 W_fc, b_fc, out, ws);
}
